// BeansBackbone_32547262169462
// MI455X (gfx1250) — compile-verified, matrix operand order checked by reading, not run on hardware
//
#include <hip/hip_runtime.h>
#include <hip/hip_bf16.h>
#include <math.h>

typedef _Float16 h4   __attribute__((ext_vector_type(4)));
typedef _Float16 v8h  __attribute__((ext_vector_type(8)));
typedef _Float16 v16h __attribute__((ext_vector_type(16)));
typedef float    v8f  __attribute__((ext_vector_type(8)));

// Problem constants (from reference)
constexpr int Bn   = 8;
constexpr int Cc   = 3;
constexpr int IMG  = 224;
constexpr int PT   = 16;
constexpr int Gg   = 14;
constexpr int Pp   = 196;   // patches
constexpr int Ss   = 197;   // tokens
constexpr int Dd   = 768;
constexpr int Hh   = 12;
constexpr int HD   = 64;
constexpr int Ll   = 12;
constexpr int Kk   = 32;    // routed keys
constexpr int QKVN = 2304;
constexpr int MLP  = 3072;

// Padded sequence so every GEMM tile is full (EXEC all-ones for WMMA)
constexpr int SP   = 208;          // 13*16
constexpr int RPAD = Bn * SP;      // 1664 = 13*128

// GEMM tiling
#define TM  128
#define TN  128
#define KC  32
#define LDH 40   // LDS row stride in halfs (80B) -> conflict-free 16B ds loads

__device__ __forceinline__ float wred_max(float v) {
    for (int o = 16; o; o >>= 1) v = fmaxf(v, __shfl_xor(v, o, 32));
    return v;
}
__device__ __forceinline__ float wred_sum(float v) {
    for (int o = 16; o; o >>= 1) v += __shfl_xor(v, o, 32);
    return v;
}

// -------------------------------------------------------------------------
// Patch gather: x (B,3,224,224) -> Xp (RPAD, 768); row b*SP+s holds patch s-1
// -------------------------------------------------------------------------
__global__ __launch_bounds__(256)
void gather_kernel(const float* __restrict__ x, float* __restrict__ Xp)
{
    int idx = blockIdx.x * 256 + threadIdx.x;       // RPAD*Dd total
    int row = idx / Dd;
    int k   = idx % Dd;
    int b = row / SP, s = row % SP;
    float val = 0.0f;
    if (s >= 1 && s <= Pp) {
        int p  = s - 1;
        int gy = p / Gg, gx = p % Gg;
        int c  = k >> 8;            // k = c*256 + py*16 + px
        int rm = k & 255;
        int py = rm >> 4, px = rm & 15;
        val = x[(((size_t)b * Cc + c) * IMG + gy * PT + py) * IMG + gx * PT + px];
    }
    Xp[(size_t)row * Dd + k] = val;
}

// -------------------------------------------------------------------------
// WMMA GEMM: out[M=RPAD x N] = A[M x Kd] @ W[N x Kd]^T + bias, fused epilogue
// MODE 0: +bias   1: +bias+resid   2: gelu(+bias)   3: embed (+bias+pos, pad->0)
// Double-buffered LDS + register prefetch: one barrier per 32-deep k-step,
// global loads for step kt+1 in flight while the 8 WMMAs of step kt execute.
// -------------------------------------------------------------------------
template <int MODE>
__global__ __launch_bounds__(256)
void gemm_wmma(const float* __restrict__ A, const float* __restrict__ W,
               const float* __restrict__ bias, const float* __restrict__ extra,
               float* __restrict__ out, int N, int Kd)
{
    __shared__ __align__(16) _Float16 As[2][TM * LDH];
    __shared__ __align__(16) _Float16 Bs[2][TN * LDH];

    const int t    = threadIdx.x;
    const int lane = t & 31;
    const int wid  = t >> 5;
    const int wm   = wid >> 1;          // 0..3 (M)
    const int wn   = wid & 1;           // 0..1 (N)
    const int l16  = lane & 15;
    const int half = lane >> 4;

    const int bM = blockIdx.y * TM;
    const int bN = blockIdx.x * TN;

    // Staging geometry: 1024 float4 slots, 4 per thread
    int srow[4], sc4[4];
    #pragma unroll
    for (int i = 0; i < 4; ++i) {
        int fid = i * 256 + t;
        srow[i] = fid >> 3;             // 0..127
        sc4[i]  = fid & 7;              // 0..7
    }

    v8f acc[2][4];
    #pragma unroll
    for (int i = 0; i < 2; ++i)
        #pragma unroll
        for (int j = 0; j < 4; ++j)
            acc[i][j] = (v8f)0.0f;

    const int ksteps = Kd / KC;

    float4 ra[4], rb[4];
    // Prefetch k-step 0
    #pragma unroll
    for (int i = 0; i < 4; ++i) {
        ra[i] = *(const float4*)(A + (size_t)(bM + srow[i]) * Kd + sc4[i] * 4);
        rb[i] = *(const float4*)(W + (size_t)(bN + srow[i]) * Kd + sc4[i] * 4);
    }
    // Stage k-step 0 into buffer 0 (f32 -> f16)
    #pragma unroll
    for (int i = 0; i < 4; ++i) {
        h4 ha, hb;
        ha.x = (_Float16)ra[i].x; ha.y = (_Float16)ra[i].y;
        ha.z = (_Float16)ra[i].z; ha.w = (_Float16)ra[i].w;
        hb.x = (_Float16)rb[i].x; hb.y = (_Float16)rb[i].y;
        hb.z = (_Float16)rb[i].z; hb.w = (_Float16)rb[i].w;
        *(h4*)(&As[0][srow[i] * LDH + sc4[i] * 4]) = ha;
        *(h4*)(&Bs[0][srow[i] * LDH + sc4[i] * 4]) = hb;
    }
    __syncthreads();

    for (int kt = 0; kt < ksteps; ++kt) {
        const int cur = kt & 1;
        const bool more = (kt + 1) < ksteps;

        // Issue global loads for next k-step first (hide under WMMA)
        if (more) {
            const int k0n = (kt + 1) * KC;
            #pragma unroll
            for (int i = 0; i < 4; ++i) {
                ra[i] = *(const float4*)(A + (size_t)(bM + srow[i]) * Kd + k0n + sc4[i] * 4);
                rb[i] = *(const float4*)(W + (size_t)(bN + srow[i]) * Kd + k0n + sc4[i] * 4);
            }
        }

        // Fragment loads per ISA 16-bit A/B layout:
        // lane holds row (A) / col (B) = l16; element e -> K = e + 8*half + 8*(e>>3)
        v16h af[2], bf[4];
        #pragma unroll
        for (int ms = 0; ms < 2; ++ms) {
            int off = (wm * 32 + ms * 16 + l16) * LDH + half * 8;
            v8h lo = *(const v8h*)(&As[cur][off]);
            v8h hi = *(const v8h*)(&As[cur][off + 16]);
            af[ms] = __builtin_shufflevector(lo, hi, 0,1,2,3,4,5,6,7,8,9,10,11,12,13,14,15);
        }
        #pragma unroll
        for (int ns = 0; ns < 4; ++ns) {
            int off = (wn * 64 + ns * 16 + l16) * LDH + half * 8;
            v8h lo = *(const v8h*)(&Bs[cur][off]);
            v8h hi = *(const v8h*)(&Bs[cur][off + 16]);
            bf[ns] = __builtin_shufflevector(lo, hi, 0,1,2,3,4,5,6,7,8,9,10,11,12,13,14,15);
        }
        #pragma unroll
        for (int ms = 0; ms < 2; ++ms)
            #pragma unroll
            for (int ns = 0; ns < 4; ++ns)
                acc[ms][ns] = __builtin_amdgcn_wmma_f32_16x16x32_f16(
                    false, af[ms], false, bf[ns], (short)0, acc[ms][ns], false, false);

        // Convert + stage next step into the other buffer, one barrier per step.
        // Buffer (kt+1)&1 was last read in step kt-1, sealed by that step's barrier.
        if (more) {
            const int nxt = cur ^ 1;
            #pragma unroll
            for (int i = 0; i < 4; ++i) {
                h4 ha, hb;
                ha.x = (_Float16)ra[i].x; ha.y = (_Float16)ra[i].y;
                ha.z = (_Float16)ra[i].z; ha.w = (_Float16)ra[i].w;
                hb.x = (_Float16)rb[i].x; hb.y = (_Float16)rb[i].y;
                hb.z = (_Float16)rb[i].z; hb.w = (_Float16)rb[i].w;
                *(h4*)(&As[nxt][srow[i] * LDH + sc4[i] * 4]) = ha;
                *(h4*)(&Bs[nxt][srow[i] * LDH + sc4[i] * 4]) = hb;
            }
            __syncthreads();
        }
    }

    // Epilogue: C/D layout -> element v maps to row (v + 8*half), col l16
    #pragma unroll
    for (int ms = 0; ms < 2; ++ms) {
        const int gr0 = bM + wm * 32 + ms * 16 + half * 8;
        const int s0  = (MODE == 3) ? (gr0 % SP) : 0;
        #pragma unroll
        for (int ns = 0; ns < 4; ++ns) {
            const int gc = bN + wn * 64 + ns * 16 + l16;
            #pragma unroll
            for (int v = 0; v < 8; ++v) {
                const int gr = gr0 + v;
                float val = acc[ms][ns][v] + bias[gc];
                if (MODE == 1) val += extra[(size_t)gr * N + gc];
                if (MODE == 2) val = 0.5f * val * (1.0f + erff(val * 0.70710678118654752f));
                if (MODE == 3) {
                    int s = s0 + v;
                    if (s >= SP) s -= SP;
                    val = (s >= 1 && s <= Pp) ? (val + extra[(size_t)s * Dd + gc]) : 0.0f;
                }
                out[(size_t)gr * N + gc] = val;
            }
        }
    }
}

// -------------------------------------------------------------------------
// cls token fixup: h[b,0,:] = cls_token + pos_embed[0]
// -------------------------------------------------------------------------
__global__ __launch_bounds__(256)
void cls_kernel(const float* __restrict__ cls_tok, const float* __restrict__ pos,
                float* __restrict__ hbuf)
{
    int idx = blockIdx.x * 256 + threadIdx.x;       // Bn*Dd
    int b = idx / Dd, n = idx % Dd;
    hbuf[(size_t)(b * SP) * Dd + n] = cls_tok[n] + pos[n];
}

// -------------------------------------------------------------------------
// LayerNorm over D=768, one block (8 waves) per row
// -------------------------------------------------------------------------
__global__ __launch_bounds__(256)
void ln_kernel(const float* __restrict__ in, const float* __restrict__ gs,
               const float* __restrict__ gb, float* __restrict__ o)
{
    const int row = blockIdx.x;
    const int t   = threadIdx.x;
    const float* x = in + (size_t)row * Dd;
    float v0 = x[t], v1 = x[t + 256], v2 = x[t + 512];
    float s1 = v0 + v1 + v2;
    float s2 = v0 * v0 + v1 * v1 + v2 * v2;
    s1 = wred_sum(s1);
    s2 = wred_sum(s2);
    __shared__ float sm1[8], sm2[8];
    if ((t & 31) == 0) { sm1[t >> 5] = s1; sm2[t >> 5] = s2; }
    __syncthreads();
    float a = 0.0f, b2 = 0.0f;
    #pragma unroll
    for (int i = 0; i < 8; ++i) { a += sm1[i]; b2 += sm2[i]; }
    float m   = a * (1.0f / Dd);
    float var = b2 * (1.0f / Dd) - m * m;
    float rs  = rsqrtf(var + 1e-5f);
    float* y = o + (size_t)row * Dd;
    y[t]       = (v0 - m) * rs * gs[t]       + gb[t];
    y[t + 256] = (v1 - m) * rs * gs[t + 256] + gb[t + 256];
    y[t + 512] = (v2 - m) * rs * gs[t + 512] + gb[t + 512];
}

// -------------------------------------------------------------------------
// Attention: one wave per (b,h,s). qkv row layout: [Q(768) K(768) V(768)],
// head-major within each. Patch queries: 32 routed keys = 1 per lane.
// -------------------------------------------------------------------------
__global__ __launch_bounds__(256)
void attn_kernel(const float* __restrict__ qkv, const int* __restrict__ routes,
                 float* __restrict__ abuf)
{
    const int wg = blockIdx.x * 8 + (threadIdx.x >> 5);
    if (wg >= Bn * Hh * Ss) return;                  // uniform per wave
    const int lane = threadIdx.x & 31;
    const int b = wg / (Hh * Ss);
    const int r = wg % (Hh * Ss);
    const int h = r / Ss;
    const int s = r % Ss;
    const size_t rowbase = (size_t)b * SP;
    const float* Q = qkv + (rowbase + s) * QKVN + h * HD;

    float o0 = 0.0f, o1 = 0.0f;
    if (s == 0) {
        // cls attends to all 197 keys: 7 chunks of 32
        float sc[7];
        #pragma unroll
        for (int c = 0; c < 7; ++c) {
            int key = c * 32 + lane;
            float d = -1e30f;
            if (key < Ss) {
                const float* Kr = qkv + (rowbase + key) * QKVN + Dd + h * HD;
                float acc = 0.0f;
                for (int dd = 0; dd < HD; ++dd) acc += Q[dd] * Kr[dd];
                d = acc * 0.125f;
            }
            sc[c] = d;
        }
        float m = -1e30f;
        #pragma unroll
        for (int c = 0; c < 7; ++c) m = fmaxf(m, sc[c]);
        m = wred_max(m);
        float pr[7], lsum = 0.0f;
        #pragma unroll
        for (int c = 0; c < 7; ++c) { pr[c] = __expf(sc[c] - m); lsum += pr[c]; }
        float inv = 1.0f / wred_sum(lsum);
        for (int c = 0; c < 7; ++c)
            for (int j = 0; j < 32; ++j) {
                int key = c * 32 + j;                // uniform
                float pj = __shfl(pr[c], j, 32) * inv;
                if (key < Ss) {
                    const float* Vr = qkv + (rowbase + key) * QKVN + 2 * Dd + h * HD;
                    o0 += pj * Vr[lane];
                    o1 += pj * Vr[lane + 32];
                }
            }
    } else {
        const int p = s - 1;
        int idx = routes[p * Kk + lane] + 1;
        const float* Kr = qkv + (rowbase + idx) * QKVN + Dd + h * HD;
        float acc = 0.0f;
        for (int dd = 0; dd < HD; ++dd) acc += Q[dd] * Kr[dd];
        float sc = acc * 0.125f;
        float m  = wred_max(sc);
        float e  = __expf(sc - m);
        float pr = e / wred_sum(e);
        for (int j = 0; j < 32; ++j) {
            float pj = __shfl(pr, j, 32);
            int   ij = __shfl(idx, j, 32);
            const float* Vr = qkv + (rowbase + ij) * QKVN + 2 * Dd + h * HD;
            o0 += pj * Vr[lane];
            o1 += pj * Vr[lane + 32];
        }
    }
    float* orow = abuf + (rowbase + s) * Dd + h * HD;
    orow[lane]      = o0;
    orow[lane + 32] = o1;
}

// -------------------------------------------------------------------------
// Unpad to (B,197,768)
// -------------------------------------------------------------------------
__global__ __launch_bounds__(256)
void outcopy_kernel(const float* __restrict__ hbuf, float* __restrict__ out)
{
    int idx = blockIdx.x * 256 + threadIdx.x;
    if (idx >= Bn * Ss * Dd) return;
    int n  = idx % Dd;
    int bs = idx / Dd;
    int b = bs / Ss, s = bs % Ss;
    out[idx] = hbuf[((size_t)(b * SP + s)) * Dd + n];
}

extern "C" void kernel_launch(void* const* d_in, const int* in_sizes, int n_in,
                              void* d_out, int out_size, void* d_ws, size_t ws_size,
                              hipStream_t stream)
{
    (void)in_sizes; (void)n_in; (void)out_size; (void)ws_size;
    const float* x       = (const float*)d_in[0];
    const float* patch_w = (const float*)d_in[1];
    const float* patch_b = (const float*)d_in[2];
    const float* cls_tok = (const float*)d_in[3];
    const float* pos     = (const float*)d_in[4];
    const float* qkv_w   = (const float*)d_in[5];
    const float* qkv_b   = (const float*)d_in[6];
    const float* proj_w  = (const float*)d_in[7];
    const float* proj_b  = (const float*)d_in[8];
    const float* ln1_s   = (const float*)d_in[9];
    const float* ln1_b   = (const float*)d_in[10];
    const float* ln2_s   = (const float*)d_in[11];
    const float* ln2_b   = (const float*)d_in[12];
    const float* fc1_w   = (const float*)d_in[13];
    const float* fc1_b   = (const float*)d_in[14];
    const float* fc2_w   = (const float*)d_in[15];
    const float* fc2_b   = (const float*)d_in[16];
    const int*   routes  = (const int*)d_in[17];
    float* out = (float*)d_out;

    // Workspace layout (floats): all activation buffers padded to RPAD rows
    float* ws = (float*)d_ws;
    const size_t RD = (size_t)RPAD * Dd;
    float* Xp   = ws;                                  // RPAD x 768
    float* hbuf = ws + RD;                             // RPAD x 768
    float* xn   = ws + 2 * RD;                         // RPAD x 768
    float* abuf = ws + 3 * RD;                         // RPAD x 768
    float* qkvb = ws + 4 * RD;                         // RPAD x 2304
    float* mbuf = ws + 4 * RD + (size_t)RPAD * QKVN;   // RPAD x 3072

    dim3 blk(256);

    // Patch embed: gather -> GEMM(+bias+pos, zero cls/pad) -> cls fixup
    gather_kernel<<<RPAD * Dd / 256, blk, 0, stream>>>(x, Xp);
    gemm_wmma<3><<<dim3(Dd / TN, RPAD / TM), blk, 0, stream>>>(
        Xp, patch_w, patch_b, pos, hbuf, Dd, Dd);
    cls_kernel<<<Bn * Dd / 256, blk, 0, stream>>>(cls_tok, pos, hbuf);

    const int attn_blocks = (Bn * Hh * Ss + 7) / 8;
    for (int l = 0; l < Ll; ++l) {
        ln_kernel<<<RPAD, blk, 0, stream>>>(hbuf, ln1_s + l * Dd, ln1_b + l * Dd, xn);
        gemm_wmma<0><<<dim3(QKVN / TN, RPAD / TM), blk, 0, stream>>>(
            xn, qkv_w + (size_t)l * QKVN * Dd, qkv_b + l * QKVN, nullptr, qkvb, QKVN, Dd);
        attn_kernel<<<attn_blocks, blk, 0, stream>>>(qkvb, routes, abuf);
        gemm_wmma<1><<<dim3(Dd / TN, RPAD / TM), blk, 0, stream>>>(
            abuf, proj_w + (size_t)l * Dd * Dd, proj_b + l * Dd, hbuf, hbuf, Dd, Dd);
        ln_kernel<<<RPAD, blk, 0, stream>>>(hbuf, ln2_s + l * Dd, ln2_b + l * Dd, xn);
        gemm_wmma<2><<<dim3(MLP / TN, RPAD / TM), blk, 0, stream>>>(
            xn, fc1_w + (size_t)l * MLP * Dd, fc1_b + l * MLP, nullptr, mbuf, MLP, Dd);
        gemm_wmma<1><<<dim3(Dd / TN, RPAD / TM), blk, 0, stream>>>(
            mbuf, fc2_w + (size_t)l * Dd * MLP, fc2_b + l * Dd, hbuf, hbuf, Dd, MLP);
    }

    outcopy_kernel<<<(Bn * Ss * Dd + 255) / 256, blk, 0, stream>>>(hbuf, out);
}
